// NeuromorphicSpikingNetwork_23579370455594
// MI455X (gfx1250) — matrix operand order repaired and checked by reading, not compiled, not run on hardware
//
#include <hip/hip_runtime.h>
#include <hip/hip_bf16.h>

typedef __attribute__((ext_vector_type(16))) _Float16 v16h;
typedef __attribute__((ext_vector_type(8)))  float    v8f;

// ---------------- constants -------------------------------------------------
#define BATCH     2048
#define TSTEPS    100
#define DECAY     0.9f
#define ADAPT     0.001f     // 0.01 * 0.1
#define TH_DECAY  0.001f
#define TH_MIN    0.8f
#define NTOT      928        // 512+256+128+32
#define KSTRIDE   520        // LDS row stride in f16 (512 + 8 pad)

// workspace layout (float units)
#define OFF_P    0           // 2048*64 bernoulli probs
#define OFF_M    131072      // 1024 membrane
#define OFF_TH   132096      // 1024 threshold
#define OFF_RC   133120      // 1024 refractory (int)
#define OFF_SM   134144      // 1024 batch-sum m
#define OFF_STH  135168      // 1024 batch-sum th
#define OFF_SRC  136192      // 1024 batch-sum rc
#define OFF_OC   137216      // 2048*32 out_count
#define OFF_WH   202752      // f16 weights region start (float units)
// f16 sub-offsets inside WH region (f16 units)
#define WH0 0
#define WH1 32768
#define WH2 163840
#define WH3 196608

// ---------------- helpers ---------------------------------------------------
__device__ __forceinline__ float rng_uniform(unsigned s) {
  s = s * 747796405u + 2891336453u;
  unsigned w = ((s >> ((s >> 28u) + 4u)) ^ s) * 277803737u;
  w = (w >> 22u) ^ w;
  return (float)(w >> 8u) * (1.0f / 16777216.0f);
}

__device__ __forceinline__ v16h load_A_lds(const _Float16* buf, int lane, int k0) {
  // A 16x32, contiguous k order used here (same as load_B_gl): lane<16 -> K = k0+0..15 ; lane>=16 -> K = k0+16..31
  const int m = lane & 15;
  const int half = lane >> 4;
  const _Float16* r = buf + m * KSTRIDE + k0 + half * 16;
  v16h a;
#pragma unroll
  for (int j = 0; j < 8; ++j) { a[j] = r[j]; a[8 + j] = r[8 + j]; }
  return a;
}

__device__ __forceinline__ v16h load_B_gl(const _Float16* Wh, int nin, int lane,
                                          int ntile, int k0) {
  // B 32x16 layout: lane%16 = N, half selects K block; B[k][n] = W[n][k]
  const int n = lane & 15;
  const int half = lane >> 4;
  const _Float16* r = Wh + (size_t)(ntile * 16 + n) * nin + k0 + half * 16;
  v16h b;
#pragma unroll
  for (int j = 0; j < 16; ++j) b[j] = r[j];
  return b;
}

// ---------------- one dense+LIF layer (per 16-row batch tile) ---------------
template <int NIN, int NOUT, bool LAST>
__device__ __forceinline__ void lif_layer(const _Float16* cur, _Float16* nxt,
                                          const _Float16* Wh,
                                          const float* m_s, const float* th_s,
                                          const int* rc_s, float* sum_m,
                                          float* sum_th, float* sum_rc,
                                          float* outcnt, int b0, int soff) {
  const int lane = threadIdx.x & 31;
  const int wave = threadIdx.x >> 5;
  const int half = lane >> 4;
  const int nlo  = lane & 15;
  constexpr int NT = NOUT / 16;

  for (int nt = wave; nt < NT; nt += 8) {
    v8f c = {};
#pragma unroll
    for (int k0 = 0; k0 < NIN; k0 += 32) {
      v16h a = load_A_lds(cur, lane, k0);
      v16h b = load_B_gl(Wh, NIN, lane, nt, k0);
      c = __builtin_amdgcn_wmma_f32_16x16x32_f16(false, a, false, b,
                                                 (short)0, c, false, false);
    }
    const int ncol = nt * 16 + nlo;
    const float mn  = m_s[soff + ncol];
    const float thn = th_s[soff + ncol];
    const int   rcn = rc_s[soff + ncol];
    const float notref = (rcn == 0) ? 1.0f : 0.0f;
    const float rcdec  = (float)((rcn > 0) ? rcn - 1 : 0);
    float sm = 0.0f, sth = 0.0f, src = 0.0f;
#pragma unroll
    for (int r = 0; r < 8; ++r) {
      const int row = r + 8 * half;            // C/D layout: M = r + 8*half
      float mp = fmaf(mn, DECAY, c[r] * notref);
      const float spike = (mp >= thn) ? 1.0f : 0.0f;
      mp *= (1.0f - spike);
      const float rcb = rcdec + spike * 2.0f;
      float thb = thn + spike * ADAPT - (thn - 1.0f) * TH_DECAY;
      thb = fmaxf(thb, TH_MIN);
      sm += mp; sth += thb; src += rcb;
      if (LAST) {
        float* oc = &outcnt[(size_t)(b0 + row) * 32 + ncol];
        *oc = *oc + spike;                     // uniquely owned: plain RMW
      } else {
        nxt[row * KSTRIDE + ncol] = (_Float16)spike;
      }
    }
    // combine lane l with l+16 (same N), then one atomic per neuron per WG
    sm  += __shfl_xor(sm, 16, 32);
    sth += __shfl_xor(sth, 16, 32);
    src += __shfl_xor(src, 16, 32);
    if (half == 0) {
      atomicAdd(&sum_m[soff + ncol], sm);
      atomicAdd(&sum_th[soff + ncol], sth);
      atomicAdd(&sum_rc[soff + ncol], src);
    }
  }
}

// ---------------- kernels ---------------------------------------------------
__global__ void snn_init(const float* __restrict__ x, const float* __restrict__ W0,
                         const float* __restrict__ W1, const float* __restrict__ W2,
                         const float* __restrict__ W3, float* ws) {
  const int tid = blockIdx.x * blockDim.x + threadIdx.x;
  const int nth = gridDim.x * blockDim.x;
  float* p      = ws + OFF_P;
  float* m_s    = ws + OFF_M;
  float* th_s   = ws + OFF_TH;
  int*   rc_s   = (int*)(ws + OFF_RC);
  float* outcnt = ws + OFF_OC;
  _Float16* Wh  = (_Float16*)(ws + OFF_WH);

  for (int i = tid; i < BATCH * 64; i += nth) {
    const int b = i >> 6, d = i & 63;
    const float xv  = x[(size_t)b * 4096 + 63 * 64 + d];   // x[b, S-1, d]
    const float lam = 0.5f / (1.0f + __expf(-xv));         // sigmoid*50/T
    p[i] = 1.0f - __expf(-lam);                            // P(poisson>=1)
  }
  for (int i = tid; i < 1024; i += nth) {
    m_s[i] = 0.0f; th_s[i] = 1.0f; rc_s[i] = 0;
    (ws + OFF_SM)[i] = 0.0f; (ws + OFF_STH)[i] = 0.0f; (ws + OFF_SRC)[i] = 0.0f;
  }
  for (int i = tid; i < BATCH * 32; i += nth) outcnt[i] = 0.0f;
  for (int i = tid; i < 512 * 64;  i += nth) Wh[WH0 + i] = (_Float16)W0[i];
  for (int i = tid; i < 256 * 512; i += nth) Wh[WH1 + i] = (_Float16)W1[i];
  for (int i = tid; i < 128 * 256; i += nth) Wh[WH2 + i] = (_Float16)W2[i];
  for (int i = tid; i < 32 * 128;  i += nth) Wh[WH3 + i] = (_Float16)W3[i];
}

__global__ void __launch_bounds__(256) snn_step(float* ws, int t) {
  __shared__ _Float16 bufA[16 * KSTRIDE];
  __shared__ _Float16 bufB[16 * KSTRIDE];

  const int b0 = blockIdx.x * 16;
  float* p      = ws + OFF_P;
  float* m_s    = ws + OFF_M;
  float* th_s   = ws + OFF_TH;
  int*   rc_s   = (int*)(ws + OFF_RC);
  float* sum_m  = ws + OFF_SM;
  float* sum_th = ws + OFF_STH;
  float* sum_rc = ws + OFF_SRC;
  float* outcnt = ws + OFF_OC;
  const _Float16* Wh = (const _Float16*)(ws + OFF_WH);

  // rate-encode input spikes for this batch tile (deterministic hash RNG)
  for (int i = threadIdx.x; i < 16 * 64; i += 256) {
    const int row = i >> 6, d = i & 63;
    const int gi  = (b0 + row) * 64 + d;
    const float u = rng_uniform((unsigned)(t * 131072 + gi) * 0x9E3779B9u + 0x7F4A7C15u);
    bufA[row * KSTRIDE + d] = (_Float16)((u < p[gi]) ? 1.0f : 0.0f);
  }
  __syncthreads();

  lif_layer<64, 512, false>(bufA, bufB, Wh + WH0, m_s, th_s, rc_s,
                            sum_m, sum_th, sum_rc, outcnt, b0, 0);
  __syncthreads();
  lif_layer<512, 256, false>(bufB, bufA, Wh + WH1, m_s, th_s, rc_s,
                             sum_m, sum_th, sum_rc, outcnt, b0, 512);
  __syncthreads();
  lif_layer<256, 128, false>(bufA, bufB, Wh + WH2, m_s, th_s, rc_s,
                             sum_m, sum_th, sum_rc, outcnt, b0, 768);
  __syncthreads();
  lif_layer<128, 32, true>(bufB, bufA, Wh + WH3, m_s, th_s, rc_s,
                           sum_m, sum_th, sum_rc, outcnt, b0, 896);
}

__global__ void snn_finalize(float* ws) {
  const int i = blockIdx.x * blockDim.x + threadIdx.x;
  if (i < NTOT) {
    const float inv = 1.0f / (float)BATCH;
    float* m_s  = ws + OFF_M;  float* th_s = ws + OFF_TH;
    int* rc_s   = (int*)(ws + OFF_RC);
    float* sm = ws + OFF_SM, *sth = ws + OFF_STH, *src = ws + OFF_SRC;
    m_s[i]  = sm[i] * inv;
    th_s[i] = sth[i] * inv;
    rc_s[i] = (int)(src[i] * inv);              // trunc like torch int cast
    sm[i] = 0.0f; sth[i] = 0.0f; src[i] = 0.0f; // ready for next step
  }
}

__global__ void snn_score(const float* __restrict__ ws_ro,
                          const float* __restrict__ Wd,
                          const float* __restrict__ bd,
                          float* __restrict__ out) {
  const int b = blockIdx.x * blockDim.x + threadIdx.x;
  if (b < BATCH) {
    const float* outcnt = ws_ro + OFF_OC;
    float s = bd[0];
#pragma unroll
    for (int n = 0; n < 32; ++n)
      s += (outcnt[(size_t)b * 32 + n] * (1.0f / (float)TSTEPS)) * Wd[n];
    out[b] = 1.0f / (1.0f + __expf(-s));
  }
}

// ---------------- launch ----------------------------------------------------
extern "C" void kernel_launch(void* const* d_in, const int* in_sizes, int n_in,
                              void* d_out, int out_size, void* d_ws, size_t ws_size,
                              hipStream_t stream) {
  const float* x  = (const float*)d_in[0];
  const float* W0 = (const float*)d_in[1];
  const float* W1 = (const float*)d_in[2];
  const float* W2 = (const float*)d_in[3];
  const float* W3 = (const float*)d_in[4];
  const float* Wd = (const float*)d_in[5];
  const float* bd = (const float*)d_in[6];
  float* ws  = (float*)d_ws;
  float* out = (float*)d_out;

  snn_init<<<256, 256, 0, stream>>>(x, W0, W1, W2, W3, ws);
  for (int t = 0; t < TSTEPS; ++t) {
    snn_step<<<BATCH / 16, 256, 0, stream>>>(ws, t);
    snn_finalize<<<4, 256, 0, stream>>>(ws);
  }
  snn_score<<<8, 256, 0, stream>>>(ws, Wd, bd, out);
}
